// AttentionPairBias_58179626992356
// MI455X (gfx1250) — matrix operand order repaired and checked by reading, not compiled, not run on hardware
//
#include <hip/hip_runtime.h>
#include <hip/hip_bf16.h>
#include <math.h>

typedef __hip_bfloat16 bf16;
typedef __attribute__((ext_vector_type(16))) __bf16 bf16x16;
typedef __attribute__((ext_vector_type(8)))  float  v8f;

union V16 { bf16x16 v; uint4 u[2]; };

#define DIMV   768
#define NSEQ   1024
#define TOKS   2048     // S*N = 2*1024
#define NH     16
#define C64    64
#define QK_SCALE 0.14433756729740643f   // 1/sqrt(48)

__device__ __forceinline__ v8f wmma_bf16(bf16x16 a, bf16x16 b, v8f c) {
  return __builtin_amdgcn_wmma_f32_16x16x32_bf16(false, a, false, b, (short)0, c, false, false);
}

// ---------------- LayerNorm of single_repr (no affine) and single_proj (scale-only) ----------
__global__ void __launch_bounds__(256) ln_kernel(const float* __restrict__ srepr,
                                                 const float* __restrict__ sproj,
                                                 const float* __restrict__ s_w,
                                                 bf16* __restrict__ aln,
                                                 bf16* __restrict__ sln) {
  __shared__ float r1[256], r2[256];
  const int tok = blockIdx.x;
  const int tid = threadIdx.x;
  const size_t base = (size_t)tok * DIMV;

  // ---- a = LN(single_repr)
  float x0 = srepr[base + tid], x1 = srepr[base + tid + 256], x2 = srepr[base + tid + 512];
  r1[tid] = x0 + x1 + x2;
  r2[tid] = x0*x0 + x1*x1 + x2*x2;
  __syncthreads();
  for (int s = 128; s > 0; s >>= 1) {
    if (tid < s) { r1[tid] += r1[tid + s]; r2[tid] += r2[tid + s]; }
    __syncthreads();
  }
  float mu = r1[0] * (1.0f / DIMV);
  float var = r2[0] * (1.0f / DIMV) - mu * mu;
  float rs = rsqrtf(var + 1e-5f);
  aln[base + tid]       = __float2bfloat16((x0 - mu) * rs);
  aln[base + tid + 256] = __float2bfloat16((x1 - mu) * rs);
  aln[base + tid + 512] = __float2bfloat16((x2 - mu) * rs);
  __syncthreads();

  // ---- s = LN(single_proj) * adaln_s_w
  x0 = sproj[base + tid]; x1 = sproj[base + tid + 256]; x2 = sproj[base + tid + 512];
  r1[tid] = x0 + x1 + x2;
  r2[tid] = x0*x0 + x1*x1 + x2*x2;
  __syncthreads();
  for (int s = 128; s > 0; s >>= 1) {
    if (tid < s) { r1[tid] += r1[tid + s]; r2[tid] += r2[tid + s]; }
    __syncthreads();
  }
  mu = r1[0] * (1.0f / DIMV);
  var = r2[0] * (1.0f / DIMV) - mu * mu;
  rs = rsqrtf(var + 1e-5f);
  sln[base + tid]       = __float2bfloat16((x0 - mu) * rs * s_w[tid]);
  sln[base + tid + 256] = __float2bfloat16((x1 - mu) * rs * s_w[tid + 256]);
  sln[base + tid + 512] = __float2bfloat16((x2 - mu) * rs * s_w[tid + 512]);
}

// ---------------- fp32 [in][out] -> bf16 [out][in] weight transpose ----------------
__global__ void __launch_bounds__(256) wt_kernel(const float* __restrict__ src,
                                                 bf16* __restrict__ dst) {
  __shared__ float tile[32][33];
  const int bo = blockIdx.x * 32;   // out base
  const int bi = blockIdx.y * 32;   // in base
  const int tx = threadIdx.x, ty = threadIdx.y;
  for (int j = ty; j < 32; j += 8)
    tile[j][tx] = src[(size_t)(bi + j) * DIMV + bo + tx];   // tile[in_l][out_l]
  __syncthreads();
  for (int j = ty; j < 32; j += 8)
    dst[(size_t)(bo + j) * DIMV + bi + tx] = __float2bfloat16(tile[tx][j]);
}

// ---------------- bf16 zero fill (q/k/vT pad) ----------------
__global__ void zero_kernel(bf16* p, int n) {
  for (int i = blockIdx.x * blockDim.x + threadIdx.x; i < n; i += gridDim.x * blockDim.x)
    p[i] = __float2bfloat16(0.0f);
}

// ---------------- WMMA GEMM cores: 16x64 output tile per wave, K=768 ----------------
__device__ __forceinline__ void gemm_core(const bf16* __restrict__ A,
                                          const bf16* __restrict__ Bt,
                                          int mbase, int nbase, int lane, v8f acc[4]) {
  const int nsel = lane & 15;
  const int acb  = (lane >> 4) << 4;   // A chunk base: 0 / 16
  const int bcb  = (lane >> 4) << 4;   // B chunk base: 0 / 16
  const size_t arow = (size_t)(mbase + nsel) * DIMV;
  for (int kb = 0; kb < DIMV; kb += 32) {
    V16 a;
    const bf16* ap = A + arow + kb + acb;
    a.u[0] = *(const uint4*)(ap);
    a.u[1] = *(const uint4*)(ap + 8);
#pragma unroll
    for (int t = 0; t < 4; ++t) {
      const bf16* bp = Bt + (size_t)(nbase + t * 16 + nsel) * DIMV + kb + bcb;
      V16 b;
      b.u[0] = *(const uint4*)(bp);
      b.u[1] = *(const uint4*)(bp + 8);
      acc[t] = wmma_bf16(a.v, b.v, acc[t]);
    }
  }
}

__device__ __forceinline__ void gemm_core2(const bf16* __restrict__ A,
                                           const bf16* __restrict__ B1,
                                           const bf16* __restrict__ B2,
                                           int mbase, int nbase, int lane,
                                           v8f acc1[4], v8f acc2[4]) {
  const int nsel = lane & 15;
  const int acb  = (lane >> 4) << 4;
  const int bcb  = (lane >> 4) << 4;
  const size_t arow = (size_t)(mbase + nsel) * DIMV;
  for (int kb = 0; kb < DIMV; kb += 32) {
    V16 a;
    const bf16* ap = A + arow + kb + acb;
    a.u[0] = *(const uint4*)(ap);
    a.u[1] = *(const uint4*)(ap + 8);
#pragma unroll
    for (int t = 0; t < 4; ++t) {
      const size_t boff = (size_t)(nbase + t * 16 + nsel) * DIMV + kb + bcb;
      V16 b;
      b.u[0] = *(const uint4*)(B1 + boff);
      b.u[1] = *(const uint4*)(B1 + boff + 8);
      acc1[t] = wmma_bf16(a.v, b.v, acc1[t]);
      b.u[0] = *(const uint4*)(B2 + boff);
      b.u[1] = *(const uint4*)(B2 + boff + 8);
      acc2[t] = wmma_bf16(a.v, b.v, acc2[t]);
    }
  }
}

// ---------------- AdaLN: a = sigmoid(s@Wg + bg)*LN(a) + s@Wskip ----------------
__global__ void __launch_bounds__(128) adaln_kernel(const bf16* __restrict__ sln,
                                                    const bf16* __restrict__ aln,
                                                    const bf16* __restrict__ wtg,
                                                    const bf16* __restrict__ wts,
                                                    const float* __restrict__ gate_b,
                                                    bf16* __restrict__ amat) {
  const int wid  = blockIdx.x * 4 + (threadIdx.x >> 5);
  const int lane = threadIdx.x & 31;
  const int mbase = (wid & 127) << 4;
  const int nbase = (wid >> 7) << 6;
  v8f ag[4], as[4];
#pragma unroll
  for (int t = 0; t < 4; ++t)
#pragma unroll
    for (int r = 0; r < 8; ++r) { ag[t][r] = 0.f; as[t][r] = 0.f; }
  gemm_core2(sln, wtg, wts, mbase, nbase, lane, ag, as);
  const int nsel = lane & 15, half = lane >> 4;
#pragma unroll
  for (int t = 0; t < 4; ++t) {
    const int col = nbase + t * 16 + nsel;
#pragma unroll
    for (int r = 0; r < 8; ++r) {
      const int row = mbase + r + half * 8;
      const float g = 1.0f / (1.0f + __expf(-(ag[t][r] + gate_b[col])));
      const float val = g * __bfloat162float(aln[(size_t)row * DIMV + col]) + as[t][r];
      amat[(size_t)row * DIMV + col] = __float2bfloat16(val);
    }
  }
}

// ---------------- QKV projections with per-head layout ----------------
// mode 0: Q [sh][tok][64] scaled  | mode 1: K [sh][tok][64] | mode 2: V^T [sh][64][tok]
__global__ void __launch_bounds__(128) qkv_kernel(const bf16* __restrict__ amat,
                                                  const bf16* __restrict__ wt,
                                                  int mode, bf16* __restrict__ dst) {
  const int wid  = blockIdx.x * 4 + (threadIdx.x >> 5);
  const int lane = threadIdx.x & 31;
  const int mbase = (wid & 127) << 4;
  const int nbase = (wid >> 7) << 6;
  v8f acc[4];
#pragma unroll
  for (int t = 0; t < 4; ++t)
#pragma unroll
    for (int r = 0; r < 8; ++r) acc[t][r] = 0.f;
  gemm_core(amat, wt, mbase, nbase, lane, acc);
  const int nsel = lane & 15, half = lane >> 4;
  const float sc = (mode == 0) ? QK_SCALE : 1.0f;
#pragma unroll
  for (int t = 0; t < 4; ++t) {
    const int col = nbase + t * 16 + nsel;
    const int h = col / 48, c = col % 48;
#pragma unroll
    for (int r = 0; r < 8; ++r) {
      const int row = mbase + r + half * 8;          // token
      const int sh = (row >> 10) * NH + h;
      const int n  = row & (NSEQ - 1);
      const bf16 v = __float2bfloat16(acc[t][r] * sc);
      if (mode == 2) dst[((size_t)sh * C64 + c) * NSEQ + n] = v;
      else           dst[((size_t)sh * NSEQ + n) * C64 + c] = v;
    }
  }
}

// ---------------- pair bias: pb[h][q][k] = (LN(pair_repr) @ w_pair)  (bandwidth-bound) -------
__global__ void __launch_bounds__(256) pair_kernel(const float* __restrict__ pr,
                                                   const float* __restrict__ lnw,
                                                   const float* __restrict__ lnb,
                                                   const float* __restrict__ wp,
                                                   float* __restrict__ pb) {
  __shared__ float wps[128 * 16];
  __shared__ float lw[128], lb[128];
  __shared__ float stage[32][16];
  const int tid = threadIdx.x;
  for (int i = tid; i < 2048; i += 256) wps[i] = wp[i];
  if (tid < 128) { lw[tid] = lnw[tid]; lb[tid] = lnb[tid]; }
  __syncthreads();

  const int qi = blockIdx.x >> 5;
  const int kbase = (blockIdx.x & 31) << 5;
  const int w = tid >> 5, lane = tid & 31;
  const int c = lane * 4;

  for (int i = 0; i < 4; ++i) {
    const int kk = kbase + w * 4 + i;
    const float4 xv = *(const float4*)(pr + ((size_t)qi * NSEQ + kk) * 128 + c);
    float s  = xv.x + xv.y + xv.z + xv.w;
    float ss = xv.x*xv.x + xv.y*xv.y + xv.z*xv.z + xv.w*xv.w;
#pragma unroll
    for (int d = 1; d < 32; d <<= 1) { s += __shfl_xor(s, d, 32); ss += __shfl_xor(ss, d, 32); }
    const float mu = s * (1.0f / 128.0f);
    const float var = ss * (1.0f / 128.0f) - mu * mu;
    const float rsq = rsqrtf(var + 1e-5f);
    const float y0 = (xv.x - mu) * rsq * lw[c]     + lb[c];
    const float y1 = (xv.y - mu) * rsq * lw[c + 1] + lb[c + 1];
    const float y2 = (xv.z - mu) * rsq * lw[c + 2] + lb[c + 2];
    const float y3 = (xv.w - mu) * rsq * lw[c + 3] + lb[c + 3];
#pragma unroll
    for (int h = 0; h < 16; ++h) {
      float p = y0 * wps[c * 16 + h] + y1 * wps[(c + 1) * 16 + h]
              + y2 * wps[(c + 2) * 16 + h] + y3 * wps[(c + 3) * 16 + h];
#pragma unroll
      for (int d = 1; d < 32; d <<= 1) p += __shfl_xor(p, d, 32);
      if (lane == h) stage[w * 4 + i][h] = p;
    }
  }
  __syncthreads();
  for (int i = tid; i < 512; i += 256) {
    const int h = i >> 5, kl = i & 31;
    pb[(size_t)h * NSEQ * NSEQ + (size_t)qi * NSEQ + kbase + kl] = stage[kl][h];
  }
}

// ---------------- flash attention: 1 wave per (s,h,16 queries), 32 keys/iter ----------------
__global__ void __launch_bounds__(128) attn_kernel(const bf16* __restrict__ q,
                                                   const bf16* __restrict__ k,
                                                   const bf16* __restrict__ vT,
                                                   const float* __restrict__ pair,
                                                   const float* __restrict__ mask,
                                                   float* __restrict__ o) {
  __shared__ bf16 plds[4][16 * 32];
  const int warp = threadIdx.x >> 5, lane = threadIdx.x & 31;
  const int wid = blockIdx.x * 4 + warp;
  const int sh = wid >> 6;                 // s*16+h
  const int qb = (wid & 63) << 4;
  const int sidx = sh >> 4, h = sh & 15;
  const bf16* qp = q  + ((size_t)sh * NSEQ + qb) * C64;
  const bf16* kp = k  + (size_t)sh * NSEQ * C64;
  const bf16* vp = vT + (size_t)sh * C64 * NSEQ;
  const float* pbb = pair + (size_t)h * NSEQ * NSEQ + (size_t)qb * NSEQ;
  float* ob = o + ((size_t)(sidx * NSEQ + qb)) * DIMV + h * 48;
  bf16* pl = plds[warp];

  const int nsel = lane & 15;
  const int half = lane >> 4;
  const int acb = half * 16;

  V16 qa0, qa1;
  {
    const bf16* r = qp + nsel * C64 + acb;
    qa0.u[0] = *(const uint4*)(r);
    qa0.u[1] = *(const uint4*)(r + 8);
    qa1.u[0] = *(const uint4*)(r + 32);
    qa1.u[1] = *(const uint4*)(r + 40);
  }

  float fm[8], fl[8];
  v8f accv[3];
#pragma unroll
  for (int r = 0; r < 8; ++r) { fm[r] = -1e30f; fl[r] = 0.f; }
#pragma unroll
  for (int f = 0; f < 3; ++f)
#pragma unroll
    for (int r = 0; r < 8; ++r) accv[f][r] = 0.f;

  for (int kb = 0; kb < NSEQ; kb += 32) {
    v8f c0, c1;
#pragma unroll
    for (int r = 0; r < 8; ++r) { c0[r] = 0.f; c1[r] = 0.f; }
    {
      const bf16* kr0 = kp + (size_t)(kb + nsel) * C64 + half * 16;
      V16 b;
      b.u[0] = *(const uint4*)(kr0);      b.u[1] = *(const uint4*)(kr0 + 8);
      c0 = wmma_bf16(qa0.v, b.v, c0);
      b.u[0] = *(const uint4*)(kr0 + 32); b.u[1] = *(const uint4*)(kr0 + 40);
      c0 = wmma_bf16(qa1.v, b.v, c0);
      const bf16* kr1 = kr0 + 16 * C64;
      b.u[0] = *(const uint4*)(kr1);      b.u[1] = *(const uint4*)(kr1 + 8);
      c1 = wmma_bf16(qa0.v, b.v, c1);
      b.u[0] = *(const uint4*)(kr1 + 32); b.u[1] = *(const uint4*)(kr1 + 40);
      c1 = wmma_bf16(qa1.v, b.v, c1);
    }
    const float mb0 = 1e8f * (mask[kb + nsel] - 1.0f);
    const float mb1 = 1e8f * (mask[kb + 16 + nsel] - 1.0f);
#pragma unroll
    for (int r = 0; r < 8; ++r) {
      const int m = r + half * 8;
      float s0 = c0[r] + mb0 + pbb[(size_t)m * NSEQ + kb + nsel];
      float s1 = c1[r] + mb1 + pbb[(size_t)m * NSEQ + kb + 16 + nsel];
      float mx = fmaxf(s0, s1);
#pragma unroll
      for (int d = 1; d < 16; d <<= 1) mx = fmaxf(mx, __shfl_xor(mx, d, 32));
      const float mn = fmaxf(fm[r], mx);
      const float al = __expf(fm[r] - mn);
      const float p0 = __expf(s0 - mn);
      const float p1 = __expf(s1 - mn);
      float rs = p0 + p1;
#pragma unroll
      for (int d = 1; d < 16; d <<= 1) rs += __shfl_xor(rs, d, 32);
      fm[r] = mn;
      fl[r] = fl[r] * al + rs;
      accv[0][r] *= al; accv[1][r] *= al; accv[2][r] *= al;
      pl[m * 32 + nsel]      = __float2bfloat16(p0);
      pl[m * 32 + 16 + nsel] = __float2bfloat16(p1);
    }
    __syncthreads();
    V16 pa;
    {
      const bf16* pr = pl + nsel * 32 + acb;
      pa.u[0] = *(const uint4*)(pr);
      pa.u[1] = *(const uint4*)(pr + 8);
    }
#pragma unroll
    for (int f = 0; f < 3; ++f) {
      const bf16* vr = vp + (size_t)(f * 16 + nsel) * NSEQ + kb + half * 16;
      V16 b;
      b.u[0] = *(const uint4*)(vr);
      b.u[1] = *(const uint4*)(vr + 8);
      accv[f] = wmma_bf16(pa.v, b.v, accv[f]);
    }
    __syncthreads();
  }
#pragma unroll
  for (int r = 0; r < 8; ++r) {
    const int m = r + half * 8;
    const float inv = 1.0f / fl[r];
    ob[(size_t)m * DIMV + nsel]      = accv[0][r] * inv;
    ob[(size_t)m * DIMV + 16 + nsel] = accv[1][r] * inv;
    ob[(size_t)m * DIMV + 32 + nsel] = accv[2][r] * inv;
  }
}

// ---------------- og = o * sigmoid(a @ Wgate + b_gate) ----------------
__global__ void __launch_bounds__(128) gate_kernel(const bf16* __restrict__ amat,
                                                   const bf16* __restrict__ wtg,
                                                   const float* __restrict__ b_gate,
                                                   const float* __restrict__ o,
                                                   bf16* __restrict__ og) {
  const int wid  = blockIdx.x * 4 + (threadIdx.x >> 5);
  const int lane = threadIdx.x & 31;
  const int mbase = (wid & 127) << 4;
  const int nbase = (wid >> 7) << 6;
  v8f acc[4];
#pragma unroll
  for (int t = 0; t < 4; ++t)
#pragma unroll
    for (int r = 0; r < 8; ++r) acc[t][r] = 0.f;
  gemm_core(amat, wtg, mbase, nbase, lane, acc);
  const int nsel = lane & 15, half = lane >> 4;
#pragma unroll
  for (int t = 0; t < 4; ++t) {
    const int col = nbase + t * 16 + nsel;
#pragma unroll
    for (int r = 0; r < 8; ++r) {
      const int row = mbase + r + half * 8;
      const float g = 1.0f / (1.0f + __expf(-(acc[t][r] + b_gate[col])));
      og[(size_t)row * DIMV + col] = __float2bfloat16(o[(size_t)row * DIMV + col] * g);
    }
  }
}

// ---------------- o2 = og @ Wo + bo ----------------
__global__ void __launch_bounds__(128) out1_kernel(const bf16* __restrict__ og,
                                                   const bf16* __restrict__ wto,
                                                   const float* __restrict__ bo,
                                                   float* __restrict__ o2f,
                                                   bf16* __restrict__ o2b) {
  const int wid  = blockIdx.x * 4 + (threadIdx.x >> 5);
  const int lane = threadIdx.x & 31;
  const int mbase = (wid & 127) << 4;
  const int nbase = (wid >> 7) << 6;
  v8f acc[4];
#pragma unroll
  for (int t = 0; t < 4; ++t)
#pragma unroll
    for (int r = 0; r < 8; ++r) acc[t][r] = 0.f;
  gemm_core(og, wto, mbase, nbase, lane, acc);
  const int nsel = lane & 15, half = lane >> 4;
#pragma unroll
  for (int t = 0; t < 4; ++t) {
    const int col = nbase + t * 16 + nsel;
#pragma unroll
    for (int r = 0; r < 8; ++r) {
      const int row = mbase + r + half * 8;
      const float val = acc[t][r] + bo[col];
      o2f[(size_t)row * DIMV + col] = val;
      o2b[(size_t)row * DIMV + col] = __float2bfloat16(val);
    }
  }
}

// ---------------- out = sigmoid(o2 @ Wout + b_out) * o2 ----------------
__global__ void __launch_bounds__(128) out2_kernel(const bf16* __restrict__ o2b,
                                                   const bf16* __restrict__ wtout,
                                                   const float* __restrict__ b_out,
                                                   const float* __restrict__ o2f,
                                                   float* __restrict__ out) {
  const int wid  = blockIdx.x * 4 + (threadIdx.x >> 5);
  const int lane = threadIdx.x & 31;
  const int mbase = (wid & 127) << 4;
  const int nbase = (wid >> 7) << 6;
  v8f acc[4];
#pragma unroll
  for (int t = 0; t < 4; ++t)
#pragma unroll
    for (int r = 0; r < 8; ++r) acc[t][r] = 0.f;
  gemm_core(o2b, wtout, mbase, nbase, lane, acc);
  const int nsel = lane & 15, half = lane >> 4;
#pragma unroll
  for (int t = 0; t < 4; ++t) {
    const int col = nbase + t * 16 + nsel;
#pragma unroll
    for (int r = 0; r < 8; ++r) {
      const int row = mbase + r + half * 8;
      const float g = 1.0f / (1.0f + __expf(-(acc[t][r] + b_out[col])));
      out[(size_t)row * DIMV + col] = g * o2f[(size_t)row * DIMV + col];
    }
  }
}

extern "C" void kernel_launch(void* const* d_in, const int* in_sizes, int n_in,
                              void* d_out, int out_size, void* d_ws, size_t ws_size,
                              hipStream_t stream) {
  (void)in_sizes; (void)n_in; (void)out_size; (void)ws_size;
  const float* single_repr  = (const float*)d_in[0];
  const float* single_proj  = (const float*)d_in[1];
  const float* pair_repr    = (const float*)d_in[2];
  const float* mask         = (const float*)d_in[3];
  const float* adaln_s_w    = (const float*)d_in[4];
  const float* adaln_gate_w = (const float*)d_in[5];
  const float* adaln_gate_b = (const float*)d_in[6];
  const float* adaln_skip_w = (const float*)d_in[7];
  const float* wq           = (const float*)d_in[8];
  const float* wk           = (const float*)d_in[9];
  const float* wv           = (const float*)d_in[10];
  const float* w_gate       = (const float*)d_in[11];
  const float* b_gate       = (const float*)d_in[12];
  const float* wo           = (const float*)d_in[13];
  const float* bo           = (const float*)d_in[14];
  const float* pair_ln_w    = (const float*)d_in[15];
  const float* pair_ln_b    = (const float*)d_in[16];
  const float* w_pair       = (const float*)d_in[17];
  const float* w_out        = (const float*)d_in[18];
  const float* b_out        = (const float*)d_in[19];
  float* out = (float*)d_out;

  // ---- workspace carve (~112 MB) ----
  char* w = (char*)d_ws;
  const size_t SZ_TOKMAT_BF = (size_t)TOKS * DIMV * sizeof(bf16);    // 3 MB
  const size_t SZ_TOKMAT_F  = (size_t)TOKS * DIMV * sizeof(float);   // 6 MB
  const size_t SZ_QKV       = (size_t)2 * NH * NSEQ * C64 * sizeof(bf16); // 4 MB
  bf16* aln  = (bf16*)w; w += SZ_TOKMAT_BF;
  bf16* sln  = (bf16*)w; w += SZ_TOKMAT_BF;
  bf16* amat = (bf16*)w; w += SZ_TOKMAT_BF;
  bf16* wt   = (bf16*)w; w += (size_t)8 * DIMV * DIMV * sizeof(bf16); // 8 transposed weights
  bf16* qbuf = (bf16*)w; w += SZ_QKV;
  bf16* kbuf = (bf16*)w; w += SZ_QKV;
  bf16* vbuf = (bf16*)w; w += SZ_QKV;
  float* pb  = (float*)w; w += (size_t)NH * NSEQ * NSEQ * sizeof(float); // 64 MB
  float* obuf = (float*)w; w += SZ_TOKMAT_F;
  bf16* og   = (bf16*)w; w += SZ_TOKMAT_BF;
  float* o2f = (float*)w; w += SZ_TOKMAT_F;
  bf16* o2b  = (bf16*)w; w += SZ_TOKMAT_BF;

  const size_t WSTEP = (size_t)DIMV * DIMV;
  bf16* wt_gate  = wt + 0 * WSTEP;
  bf16* wt_skip  = wt + 1 * WSTEP;
  bf16* wt_q     = wt + 2 * WSTEP;
  bf16* wt_k     = wt + 3 * WSTEP;
  bf16* wt_v     = wt + 4 * WSTEP;
  bf16* wt_gate2 = wt + 5 * WSTEP;
  bf16* wt_o     = wt + 6 * WSTEP;
  bf16* wt_out   = wt + 7 * WSTEP;

  // 1) layernorms
  ln_kernel<<<TOKS, 256, 0, stream>>>(single_repr, single_proj, adaln_s_w, aln, sln);

  // 2) transpose+convert the eight 768x768 weights
  dim3 tg(24, 24), tb(32, 8);
  wt_kernel<<<tg, tb, 0, stream>>>(adaln_gate_w, wt_gate);
  wt_kernel<<<tg, tb, 0, stream>>>(adaln_skip_w, wt_skip);
  wt_kernel<<<tg, tb, 0, stream>>>(wq, wt_q);
  wt_kernel<<<tg, tb, 0, stream>>>(wk, wt_k);
  wt_kernel<<<tg, tb, 0, stream>>>(wv, wt_v);
  wt_kernel<<<tg, tb, 0, stream>>>(w_gate, wt_gate2);
  wt_kernel<<<tg, tb, 0, stream>>>(wo, wt_o);
  wt_kernel<<<tg, tb, 0, stream>>>(w_out, wt_out);

  // 3) zero q/k/vT (for the 48->64 head-dim padding); buffers are contiguous
  {
    const int n = (int)(3 * SZ_QKV / sizeof(bf16));
    zero_kernel<<<2048, 256, 0, stream>>>(qbuf, n);
  }

  const int GEMM_BLOCKS = (128 * 12) / 4;   // 1536 wave tiles / 4 waves per block

  // 4) AdaLN fused dual GEMM
  adaln_kernel<<<GEMM_BLOCKS, 128, 0, stream>>>(sln, aln, wt_gate, wt_skip, adaln_gate_b, amat);

  // 5) Q/K/V projections
  qkv_kernel<<<GEMM_BLOCKS, 128, 0, stream>>>(amat, wt_q, 0, qbuf);
  qkv_kernel<<<GEMM_BLOCKS, 128, 0, stream>>>(amat, wt_k, 1, kbuf);
  qkv_kernel<<<GEMM_BLOCKS, 128, 0, stream>>>(amat, wt_v, 2, vbuf);

  // 6) pair bias (streams 512 MB of pair_repr once — the bandwidth floor)
  pair_kernel<<<NSEQ * 32, 256, 0, stream>>>(pair_repr, pair_ln_w, pair_ln_b, w_pair, pb);

  // 7) flash attention
  attn_kernel<<<(2 * NH * (NSEQ / 16)) / 4, 128, 0, stream>>>(qbuf, kbuf, vbuf, pb, mask, obuf);

  // 8) output gating + projections
  gate_kernel<<<GEMM_BLOCKS, 128, 0, stream>>>(amat, wt_gate2, b_gate, obuf, og);
  out1_kernel<<<GEMM_BLOCKS, 128, 0, stream>>>(og, wt_o, bo, o2f, o2b);
  out2_kernel<<<GEMM_BLOCKS, 128, 0, stream>>>(o2b, wt_out, b_out, o2f, out);
}
